// TransformerEncoder_42855183679935
// MI455X (gfx1250) — matrix operand order repaired and checked by reading, not compiled, not run on hardware
//
#include <hip/hip_runtime.h>
#include <math.h>

// ---------------- problem constants ----------------
#define LAYERS  2
#define BATCH   4
#define CH      256
#define HH      128
#define WW      128
#define DD      128          // CH/2
#define NHEADS  2
#define BHD     (BATCH*NHEADS)   // 8
#define HWSZ    (HH*WW)          // 16384
#define HK      16
#define WK      16
#define NS      (HK*WK)          // 256
#define KSZ     9
#define CSTRIDE 8
#define CPAD    4
#define ATT_SCALE 0.17677669529663687f   // 32^-0.5

#define LDSPITCH 40   // padded row pitch (elements) for LDS tiles

typedef __attribute__((ext_vector_type(16))) __bf16 v16bf;
typedef __attribute__((ext_vector_type(8)))  float  v8f;

// ---------------- bf16 split helpers (hi/lo decomposition) ------------------
__device__ __forceinline__ void split_bf16(float f, unsigned short& h, unsigned short& l) {
    union { __bf16 b; unsigned short u; } c;
    c.b = (__bf16)f;                 // RNE high part
    h = c.u;
    union { unsigned short u; __bf16 b; } d; d.u = h;
    const float fh = (float)d.b;
    union { __bf16 b; unsigned short u; } e;
    e.b = (__bf16)(f - fh);          // residual
    l = e.u;
}

__device__ __forceinline__ v16bf frag16(const unsigned short* p0, const unsigned short* p1) {
    union { uint4 q[2]; v16bf v; } u;
    u.q[0] = *(const uint4*)p0;      // elements 0..7  (VGPRs 0..3)
    u.q[1] = *(const uint4*)p1;      // elements 8..15 (VGPRs 4..7)
    return u.v;
}

// ---------------- cooperative tile staging (global -> LDS, fp32 -> bf16x2) --
// A tile: 128 x 32 (M x K), B tile: 32 x 64 (K x N), stored as [n][k] for B.
template <bool ACONTIG, bool BCONTIG>
__device__ __forceinline__ void stage_tiles(const float* __restrict__ Ap,
                                            const float* __restrict__ Bp,
                                            long rsA, long csA, long rsB, long csB,
                                            long mBlk, long nBlk, int k0, float alpha, int t,
                                            unsigned short* Ah, unsigned short* Al,
                                            unsigned short* Bh, unsigned short* Bl)
{
    if constexpr (ACONTIG) {
        // contiguous-K rows: vector loads (rsA is a multiple of 4 in all such calls)
        const int r  = t >> 1;
        const int c0 = (t & 1) * 16;
        const float* src = Ap + (mBlk + r) * rsA + (long)(k0 + c0);
#pragma unroll
        for (int i = 0; i < 16; i += 4) {
            const float4 v4 = *(const float4*)(src + i);
            const float vv[4] = {v4.x, v4.y, v4.z, v4.w};
#pragma unroll
            for (int j = 0; j < 4; ++j) {
                unsigned short h, l;
                split_bf16(alpha * vv[j], h, l);
                Ah[r * LDSPITCH + c0 + i + j] = h;
                Al[r * LDSPITCH + c0 + i + j] = l;
            }
        }
    } else {
        // strided-K (rsA==1 in these calls): coalesce along M
        const int m = t & 127;
        const int g = t >> 7;
#pragma unroll
        for (int i = 0; i < 16; ++i) {
            const int c = g * 16 + i;
            unsigned short h, l;
            split_bf16(alpha * Ap[(mBlk + m) * rsA + (long)(k0 + c) * csA], h, l);
            Ah[m * LDSPITCH + c] = h;
            Al[m * LDSPITCH + c] = l;
        }
    }
    if constexpr (BCONTIG) {
        // coalesce along N
        const int n  = t & 63;
        const int kb = (t >> 6) * 8;
#pragma unroll
        for (int i = 0; i < 8; ++i) {
            const int k = kb + i;
            unsigned short h, l;
            split_bf16(Bp[(long)(k0 + k) * rsB + (nBlk + n) * csB], h, l);
            Bh[n * LDSPITCH + k] = h;
            Bl[n * LDSPITCH + k] = l;
        }
    } else {
        // rsB==1 in these calls: coalesce along K
        const int k  = t & 31;
        const int nb = (t >> 5) * 8;
#pragma unroll
        for (int i = 0; i < 8; ++i) {
            const int n = nb + i;
            unsigned short h, l;
            split_bf16(Bp[(long)(k0 + k) * rsB + (long)(nBlk + n) * csB], h, l);
            Bh[n * LDSPITCH + k] = h;
            Bl[n * LDSPITCH + k] = l;
        }
    }
}

// prefetch the addresses this thread will stage in the next K-chunk
template <bool ACONTIG, bool BCONTIG>
__device__ __forceinline__ void prefetch_next(const float* __restrict__ Ap,
                                              const float* __restrict__ Bp,
                                              long rsA, long csA, long rsB, long csB,
                                              long mBlk, long nBlk, int k1, int t)
{
    if constexpr (ACONTIG)
        __builtin_prefetch(Ap + (mBlk + (t >> 1)) * rsA + (long)(k1 + (t & 1) * 16), 0, 1);
    else
        __builtin_prefetch(Ap + (mBlk + (t & 127)) * rsA + (long)(k1 + (t >> 7) * 16) * csA, 0, 1);
    if constexpr (BCONTIG)
        __builtin_prefetch(Bp + (long)(k1 + (t >> 6) * 8) * rsB + (nBlk + (t & 63)) * csB, 0, 1);
    else
        __builtin_prefetch(Bp + (long)(k1 + (t & 31)) * rsB + (long)(nBlk + (t >> 5) * 8) * csB, 0, 1);
}

// ---------------- generic batched strided GEMM via bf16-split WMMA ----------
// C[z][m,n] = relu?( a1*(A1 B1) + a2*(A2 B2) + bias[m*bsM + n*bsN] )
// Block tile: 128(M) x 64(N); wave w owns rows [16w,16w+16), all 64 columns.
// Requirements: M%128==0, N%64==0, K%32==0 (true for every call site).
template <bool DUAL, bool ACONTIG, bool BCONTIG>
__global__ void __launch_bounds__(256)
wmma_gemm(const float* __restrict__ A1, const float* __restrict__ B1, float alpha1,
          const float* __restrict__ A2, const float* __restrict__ B2, float alpha2,
          const float* __restrict__ bias, long bsM, long bsN,
          float* __restrict__ Cd, int K, int relu,
          long rsA, long csA, long bA,
          long rsB, long csB, long bB,
          long rsC, long csC, long bC)
{
    __shared__ __align__(16) unsigned short As1h[128 * LDSPITCH];
    __shared__ __align__(16) unsigned short As1l[128 * LDSPITCH];
    __shared__ __align__(16) unsigned short Bs1h[64 * LDSPITCH];
    __shared__ __align__(16) unsigned short Bs1l[64 * LDSPITCH];
    __shared__ __align__(16) unsigned short As2h[DUAL ? 128 * LDSPITCH : 8];
    __shared__ __align__(16) unsigned short As2l[DUAL ? 128 * LDSPITCH : 8];
    __shared__ __align__(16) unsigned short Bs2h[DUAL ? 64 * LDSPITCH : 8];
    __shared__ __align__(16) unsigned short Bs2l[DUAL ? 64 * LDSPITCH : 8];

    const int  t    = threadIdx.x;
    const int  lane = t & 31;
    const int  wave = t >> 5;
    const int  half = lane >> 4;
    const int  nl   = lane & 15;
    const long mBlk = (long)blockIdx.y * 128;
    const long nBlk = (long)blockIdx.x * 64;
    const long z    = blockIdx.z;

    const float* A1p = A1 + z * bA;
    const float* B1p = B1 + z * bB;
    const float* A2p = DUAL ? (A2 + z * bA) : nullptr;
    const float* B2p = DUAL ? (B2 + z * bB) : nullptr;
    float*       Cp  = Cd + z * bC;

    v8f acc[4];
#pragma unroll
    for (int tt = 0; tt < 4; ++tt) acc[tt] = (v8f){0.f,0.f,0.f,0.f,0.f,0.f,0.f,0.f};

    for (int k0 = 0; k0 < K; k0 += 32) {
        stage_tiles<ACONTIG, BCONTIG>(A1p, B1p, rsA, csA, rsB, csB, mBlk, nBlk, k0, alpha1, t,
                                      As1h, As1l, Bs1h, Bs1l);
        if constexpr (DUAL)
            stage_tiles<ACONTIG, BCONTIG>(A2p, B2p, rsA, csA, rsB, csB, mBlk, nBlk, k0, alpha2, t,
                                          As2h, As2l, Bs2h, Bs2l);
        if (k0 + 32 < K) {
            prefetch_next<ACONTIG, BCONTIG>(A1p, B1p, rsA, csA, rsB, csB, mBlk, nBlk, k0 + 32, t);
            if constexpr (DUAL)
                prefetch_next<ACONTIG, BCONTIG>(A2p, B2p, rsA, csA, rsB, csB, mBlk, nBlk, k0 + 32, t);
        }
        __syncthreads();

        // A fragments (one per wave, reused for 4 N-tiles)
        const unsigned short* ar1h = As1h + (wave * 16 + nl) * LDSPITCH;
        const unsigned short* ar1l = As1l + (wave * 16 + nl) * LDSPITCH;
        const v16bf a1h = frag16(ar1h + 8 * half, ar1h + 16 + 8 * half);
        const v16bf a1l = frag16(ar1l + 8 * half, ar1l + 16 + 8 * half);
        v16bf a2h, a2l;
        if constexpr (DUAL) {
            const unsigned short* ar2h = As2h + (wave * 16 + nl) * LDSPITCH;
            const unsigned short* ar2l = As2l + (wave * 16 + nl) * LDSPITCH;
            a2h = frag16(ar2h + 8 * half, ar2h + 16 + 8 * half);
            a2l = frag16(ar2l + 8 * half, ar2l + 16 + 8 * half);
        }

#pragma unroll
        for (int tt = 0; tt < 4; ++tt) {
            const unsigned short* br1h = Bs1h + (tt * 16 + nl) * LDSPITCH + 8 * half;
            const unsigned short* br1l = Bs1l + (tt * 16 + nl) * LDSPITCH + 8 * half;
            const v16bf b1h = frag16(br1h, br1h + 16);
            const v16bf b1l = frag16(br1l, br1l + 16);
            acc[tt] = __builtin_amdgcn_wmma_f32_16x16x32_bf16(false, a1h, false, b1h, (short)0, acc[tt], false, false);
            acc[tt] = __builtin_amdgcn_wmma_f32_16x16x32_bf16(false, a1h, false, b1l, (short)0, acc[tt], false, false);
            acc[tt] = __builtin_amdgcn_wmma_f32_16x16x32_bf16(false, a1l, false, b1h, (short)0, acc[tt], false, false);
            if constexpr (DUAL) {
                const unsigned short* br2h = Bs2h + (tt * 16 + nl) * LDSPITCH + 8 * half;
                const unsigned short* br2l = Bs2l + (tt * 16 + nl) * LDSPITCH + 8 * half;
                const v16bf b2h = frag16(br2h, br2h + 16);
                const v16bf b2l = frag16(br2l, br2l + 16);
                acc[tt] = __builtin_amdgcn_wmma_f32_16x16x32_bf16(false, a2h, false, b2h, (short)0, acc[tt], false, false);
                acc[tt] = __builtin_amdgcn_wmma_f32_16x16x32_bf16(false, a2h, false, b2l, (short)0, acc[tt], false, false);
                acc[tt] = __builtin_amdgcn_wmma_f32_16x16x32_bf16(false, a2l, false, b2h, (short)0, acc[tt], false, false);
            }
        }
        __syncthreads();
    }

#pragma unroll
    for (int tt = 0; tt < 4; ++tt) {
#pragma unroll
        for (int i = 0; i < 8; ++i) {
            const long m = mBlk + wave * 16 + i + 8 * half;
            const long n = nBlk + tt * 16 + nl;
            float v = acc[tt][i];
            if (bias) v += bias[m * bsM + n * bsN];
            if (relu) v = fmaxf(v, 0.f);
            Cp[m * rsC + n * csC] = v;
        }
    }
}

// ---------------- block reductions ----------------
template <int NT>
__device__ __forceinline__ float blockSum(float v, float* red) {
    red[threadIdx.x] = v;
    __syncthreads();
    for (int s = NT >> 1; s > 0; s >>= 1) {
        if ((int)threadIdx.x < s) red[threadIdx.x] += red[threadIdx.x + s];
        __syncthreads();
    }
    float r = red[0];
    __syncthreads();
    return r;
}
template <int NT>
__device__ __forceinline__ float blockMax(float v, float* red) {
    red[threadIdx.x] = v;
    __syncthreads();
    for (int s = NT >> 1; s > 0; s >>= 1) {
        if ((int)threadIdx.x < s) red[threadIdx.x] = fmaxf(red[threadIdx.x], red[threadIdx.x + s]);
        __syncthreads();
    }
    float r = red[0];
    __syncthreads();
    return r;
}

// ---------------- depthwise conv + channel-LN + GELU + pointwise -> offsets -
__global__ void __launch_bounds__(256)
conv_offset_kernel(const float* __restrict__ q,
                   const float* __restrict__ dw_w, const float* __restrict__ dw_b,
                   const float* __restrict__ ln_g, const float* __restrict__ ln_b,
                   const float* __restrict__ pw_w,
                   float* __restrict__ offp)
{
    __shared__ float red[256];
    const int c   = threadIdx.x;
    const int idx = blockIdx.x;          // b*NS + s
    const int s   = idx % NS;
    const int b   = idx / NS;
    const int hk  = s / WK, wk = s % WK;

    const float* qc = q + ((long)b * CH + c) * HWSZ;
    const float* wc = dw_w + (long)c * KSZ * KSZ;
    float acc = dw_b[c];
    const int h0 = hk * CSTRIDE - CPAD;
    const int w0 = wk * CSTRIDE - CPAD;
#pragma unroll
    for (int i = 0; i < KSZ; ++i) {
        const int hh = h0 + i;
        if (hh < 0 || hh >= HH) continue;
#pragma unroll
        for (int j = 0; j < KSZ; ++j) {
            const int ww2 = w0 + j;
            if (ww2 < 0 || ww2 >= WW) continue;
            acc += qc[hh * WW + ww2] * wc[i * KSZ + j];
        }
    }
    const float mean = blockSum<256>(acc, red) * (1.0f / CH);
    const float yc   = acc - mean;
    const float var  = blockSum<256>(yc * yc, red) * (1.0f / CH);
    const float yn   = yc * rsqrtf(var + 1e-5f) * ln_g[c] + ln_b[c];
    const float g    = 0.5f * yn * (1.0f + erff(yn * 0.70710678118654752f));
    const float p0   = blockSum<256>(g * pw_w[c], red);
    const float p1   = blockSum<256>(g * pw_w[CH + c], red);
    if (c == 0) { offp[idx * 2 + 0] = p0; offp[idx * 2 + 1] = p1; }
}

// ---------------- bilinear (zero padded, clamped gather) --------------------
__device__ __forceinline__ float bilin_zero(const float* __restrict__ img,
                                            float yv, float xv, int Hi, int Wi)
{
    const float x  = (xv + 1.0f) * 0.5f * (float)(Wi - 1);
    const float y  = (yv + 1.0f) * 0.5f * (float)(Hi - 1);
    const float x0 = floorf(x), y0 = floorf(y);
    const float wx1 = x - x0, wx0 = 1.f - wx1;
    const float wy1 = y - y0, wy0 = 1.f - wy1;
    float out = 0.f;
#pragma unroll
    for (int dy = 0; dy < 2; ++dy) {
#pragma unroll
        for (int dx = 0; dx < 2; ++dx) {
            const float yy = y0 + dy, xx = x0 + dx;
            const bool valid = (xx >= 0.f) && (xx <= (float)(Wi - 1)) &&
                               (yy >= 0.f) && (yy <= (float)(Hi - 1));
            const int xi = (int)fminf(fmaxf(xx, 0.f), (float)(Wi - 1));
            const int yi = (int)fminf(fmaxf(yy, 0.f), (float)(Hi - 1));
            const float wgt = (dy ? wy1 : wy0) * (dx ? wx1 : wx0);
            out += img[yi * Wi + xi] * (valid ? wgt : 0.f);
        }
    }
    return out;
}

// ---------------- deformable sampling of x: xs[b,c,s] -----------------------
__global__ void __launch_bounds__(256)
grid_sample_kernel(const float* __restrict__ x, const float* __restrict__ offp,
                   float* __restrict__ xs)
{
    const int gid = blockIdx.x * 256 + threadIdx.x;
    const int s = gid % NS;
    const int c = (gid / NS) % CH;
    const int b = gid / (NS * CH);
    const int hk = s / WK, wk = s % WK;
    const float ry = (0.5f + hk) / (float)(HK - 1) * 2.0f - 1.0f;
    const float rx = (0.5f + wk) / (float)(WK - 1) * 2.0f - 1.0f;
    const float py = offp[(b * NS + s) * 2 + 0] + ry;
    const float px = offp[(b * NS + s) * 2 + 1] + rx;
    const float* img = x + ((long)b * CH + c) * HWSZ;
    xs[gid] = bilin_zero(img, py, px, HH, WW);
}

// ---------------- RPE bias + softmax over n (one block per (bh,m)) ----------
__global__ void __launch_bounds__(256)
softmax_bias_kernel(float* __restrict__ attn, const float* __restrict__ offp,
                    const float* __restrict__ rpe)
{
    __shared__ float red[256];
    const int  n  = threadIdx.x;
    const long blk = blockIdx.x;                 // bh*HWSZ + m
    const int  m  = (int)(blk % HWSZ);
    const int  bh = (int)(blk / HWSZ);
    const int  b  = bh / NHEADS, nh = bh % NHEADS;

    const int hk = n / WK, wk = n % WK;
    const float ry = (0.5f + hk) / (float)(HK - 1) * 2.0f - 1.0f;
    const float rx = (0.5f + wk) / (float)(WK - 1) * 2.0f - 1.0f;
    const float py = offp[(b * NS + n) * 2 + 0] + ry;
    const float px = offp[(b * NS + n) * 2 + 1] + rx;

    const int h = m / WW, w = m % WW;
    const float qgy = (float)h / (float)(HH - 1) * 2.0f - 1.0f;
    const float qgx = (float)w / (float)(WW - 1) * 2.0f - 1.0f;
    const float dy = (qgy - py) * 0.5f;
    const float dx = (qgx - px) * 0.5f;

    const float* img = rpe + (long)nh * (2 * HH - 1) * (2 * WW - 1);
    const float biasv = bilin_zero(img, dy, dx, 2 * HH - 1, 2 * WW - 1);

    const long base = blk * NS;
    const float logit = attn[base + n] + biasv;
    const float mx = blockMax<256>(logit, red);
    const float e  = expf(logit - mx);
    const float ss = blockSum<256>(e, red);
    attn[base + n] = e / ss;
}

// ---------------- complex layernorm (+residual, strided in/out) -------------
__global__ void __launch_bounds__(128)
cln_kernel(const float* __restrict__ a_in, const float* __restrict__ b_in,
           const float* __restrict__ res_a, const float* __restrict__ res_b, int res_bchw,
           const float* __restrict__ grr, const float* __restrict__ gri,
           const float* __restrict__ gii, const float* __restrict__ brp,
           const float* __restrict__ bip,
           float* __restrict__ out_a, float* __restrict__ out_b, int out_bchw)
{
    __shared__ float red[128];
    const int  d   = threadIdx.x;
    const long row = blockIdx.x;            // m*BHD + bh
    const int  bh  = (int)(row % BHD);
    const long m   = row / BHD;

    float a = a_in[row * DD + d];
    float b = b_in[row * DD + d];
    const float ma = blockSum<128>(a, red) * (1.f / DD);
    const float mb = blockSum<128>(b, red) * (1.f / DD);
    a -= ma; b -= mb;
    const float vrr = blockSum<128>(a * a, red) * (1.f / DD) + 1e-20f;
    const float vii = blockSum<128>(b * b, red) * (1.f / DD) + 1e-20f;
    const float vri = blockSum<128>(a * b, red) * (1.f / DD);
    const float s   = sqrtf(vrr * vii - vri * vri);
    const float t   = sqrtf(vrr + vii + 2.f * s);
    const float inv = 1.f / (s * t);
    const float wrr = (vii + s) * inv;
    const float wii = (vrr + s) * inv;
    const float wri = -vri * inv;
    const float na = wrr * a + wri * b;
    const float nb = wri * a + wii * b;
    float oa = grr[d] * na + gri[d] * nb + brp[d];
    float ob = gri[d] * na + gii[d] * nb + bip[d];

    const long ridx = res_bchw ? (((long)bh * DD + d) * HWSZ + m) : (row * DD + d);
    oa += res_a[ridx];
    ob += res_b[ridx];
    const long oidx = out_bchw ? (((long)bh * DD + d) * HWSZ + m) : (row * DD + d);
    out_a[oidx] = oa;
    out_b[oidx] = ob;
}

// ---------------- host-side GEMM wrapper (template dispatch) ----------------
static inline void launch_gemm(hipStream_t st, int M, int N, int K, int batch,
                               const float* A1, long rsA, long csA, long bA,
                               const float* B1, long rsB, long csB, long bB,
                               float alpha1,
                               const float* A2, const float* B2, float alpha2,
                               const float* bias, long bsM, long bsN,
                               float* Cd, long rsC, long csC, long bC, int relu)
{
    dim3 grid(N / 64, M / 128, batch);
    const bool dual = (A2 != nullptr);
    const bool ac   = (csA == 1);
    const bool bc   = (csB == 1);
#define WG_ARGS A1, B1, alpha1, A2, B2, alpha2, bias, bsM, bsN, Cd, K, relu, \
                rsA, csA, bA, rsB, csB, bB, rsC, csC, bC
    if (dual) {
        if (ac) { if (bc) wmma_gemm<true , true , true ><<<grid, 256, 0, st>>>(WG_ARGS);
                  else    wmma_gemm<true , true , false><<<grid, 256, 0, st>>>(WG_ARGS); }
        else    { if (bc) wmma_gemm<true , false, true ><<<grid, 256, 0, st>>>(WG_ARGS);
                  else    wmma_gemm<true , false, false><<<grid, 256, 0, st>>>(WG_ARGS); }
    } else {
        if (ac) { if (bc) wmma_gemm<false, true , true ><<<grid, 256, 0, st>>>(WG_ARGS);
                  else    wmma_gemm<false, true , false><<<grid, 256, 0, st>>>(WG_ARGS); }
        else    { if (bc) wmma_gemm<false, false, true ><<<grid, 256, 0, st>>>(WG_ARGS);
                  else    wmma_gemm<false, false, false><<<grid, 256, 0, st>>>(WG_ARGS); }
    }
#undef WG_ARGS
}

extern "C" void kernel_launch(void* const* d_in, const int* in_sizes, int n_in,
                              void* d_out, int out_size, void* d_ws, size_t ws_size,
                              hipStream_t stream) {
    (void)in_sizes; (void)n_in; (void)out_size; (void)ws_size;
    const float* xA0      = (const float*)d_in[0];
    const float* xB0      = (const float*)d_in[1];
    const float* wq       = (const float*)d_in[2];
    const float* bq       = (const float*)d_in[3];
    const float* wk       = (const float*)d_in[4];
    const float* bk       = (const float*)d_in[5];
    const float* wv       = (const float*)d_in[6];
    const float* bv       = (const float*)d_in[7];
    const float* off_dw_w = (const float*)d_in[8];
    const float* off_dw_b = (const float*)d_in[9];
    const float* off_ln_g = (const float*)d_in[10];
    const float* off_ln_b = (const float*)d_in[11];
    const float* off_pw_w = (const float*)d_in[12];
    const float* fc1_wr   = (const float*)d_in[13];
    const float* fc1_wi   = (const float*)d_in[14];
    const float* fc1_br   = (const float*)d_in[15];
    const float* fc1_bi   = (const float*)d_in[16];
    const float* fc2_wr   = (const float*)d_in[17];
    const float* fc2_wi   = (const float*)d_in[18];
    const float* fc2_br   = (const float*)d_in[19];
    const float* fc2_bi   = (const float*)d_in[20];
    const float* ln_grr   = (const float*)d_in[21];
    const float* ln_gri   = (const float*)d_in[22];
    const float* ln_gii   = (const float*)d_in[23];
    const float* ln_brA   = (const float*)d_in[24];
    const float* ln_biA   = (const float*)d_in[25];
    const float* rpe      = (const float*)d_in[26];

    const size_t XSZ   = (size_t)BATCH * CH * HWSZ;      // 16.78M floats
    const size_t ATTSZ = (size_t)BHD * HWSZ * NS;        // 33.55M floats == 2*XSZ

    char* ws = (char*)d_ws;
    size_t cur = 0;
    auto alloc = [&](size_t nfloats) -> float* {
        float* p = (float*)(ws + cur);
        cur += ((nfloats * sizeof(float)) + 255) & ~(size_t)255;
        return p;
    };
    float* bufXA  = alloc(XSZ);                 // layer-0 output A
    float* bufXB  = alloc(XSZ);                 // layer-0 output B
    float* bufQA  = alloc(XSZ);                 // qa, then oa, then fc2-out A
    float* bufQB  = alloc(XSZ);                 // qb, then ob, then fc2-out B
    float* bufATT = alloc(ATTSZ);               // attn, then xa2|xb2
    float* bufHA  = alloc(XSZ);                 // fc1 out A
    float* bufHB  = alloc(XSZ);                 // fc1 out B
    float* bufXSA = alloc((size_t)BATCH * CH * NS);
    float* bufXSB = alloc((size_t)BATCH * CH * NS);
    float* bufKA  = alloc((size_t)BHD * DD * NS);
    float* bufKB  = alloc((size_t)BHD * DD * NS);
    float* bufVA  = alloc((size_t)BHD * DD * NS);
    float* bufVB  = alloc((size_t)BHD * DD * NS);
    float* bufOfA = alloc((size_t)BATCH * NS * 2);
    float* bufOfB = alloc((size_t)BATCH * NS * 2);
    float* xa2p = bufATT;          // reuse attn region after it is consumed
    float* xb2p = bufATT + XSZ;

    for (int l = 0; l < LAYERS; ++l) {
        const float* xa = (l == 0) ? xA0 : bufXA;
        const float* xb = (l == 0) ? xB0 : bufXB;
        float* outA = (l == LAYERS - 1) ? (float*)d_out : bufXA;
        float* outB = (l == LAYERS - 1) ? ((float*)d_out + XSZ) : bufXB;

        const float* wq_l  = wq + (size_t)l * CH * CH;
        const float* bq_l  = bq + (size_t)l * CH;
        const float* wk_l  = wk + (size_t)l * CH * CH;
        const float* bk_l  = bk + (size_t)l * CH;
        const float* wv_l  = wv + (size_t)l * CH * CH;
        const float* bv_l  = bv + (size_t)l * CH;
        const float* dww_l = off_dw_w + (size_t)l * CH * KSZ * KSZ;
        const float* dwb_l = off_dw_b + (size_t)l * CH;
        const float* lng_l = off_ln_g + (size_t)l * CH;
        const float* lnb_l = off_ln_b + (size_t)l * CH;
        const float* pww_l = off_pw_w + (size_t)l * 2 * CH;
        const float* rpe_l = rpe + (size_t)l * NHEADS * (2 * HH - 1) * (2 * WW - 1);

        // ---- Q projections: q[b,o,hw] = wq@x + bq ----
        launch_gemm(stream, CH, HWSZ, CH, BATCH,
                    wq_l, CH, 1, 0,
                    xa, HWSZ, 1, (long)CH * HWSZ, 1.f,
                    nullptr, nullptr, 0.f,
                    bq_l, 1, 0,
                    bufQA, HWSZ, 1, (long)CH * HWSZ, 0);
        launch_gemm(stream, CH, HWSZ, CH, BATCH,
                    wq_l, CH, 1, 0,
                    xb, HWSZ, 1, (long)CH * HWSZ, 1.f,
                    nullptr, nullptr, 0.f,
                    bq_l, 1, 0,
                    bufQB, HWSZ, 1, (long)CH * HWSZ, 0);

        // ---- offset head ----
        conv_offset_kernel<<<BATCH * NS, 256, 0, stream>>>(bufQA, dww_l, dwb_l, lng_l, lnb_l, pww_l, bufOfA);
        conv_offset_kernel<<<BATCH * NS, 256, 0, stream>>>(bufQB, dww_l, dwb_l, lng_l, lnb_l, pww_l, bufOfB);

        // ---- deformable sampling ----
        grid_sample_kernel<<<(BATCH * CH * NS) / 256, 256, 0, stream>>>(xa, bufOfA, bufXSA);
        grid_sample_kernel<<<(BATCH * CH * NS) / 256, 256, 0, stream>>>(xb, bufOfB, bufXSB);

        // ---- K/V projections on sampled points ----
        launch_gemm(stream, CH, NS, CH, BATCH, wk_l, CH, 1, 0,
                    bufXSA, NS, 1, (long)CH * NS, 1.f, nullptr, nullptr, 0.f,
                    bk_l, 1, 0, bufKA, NS, 1, (long)CH * NS, 0);
        launch_gemm(stream, CH, NS, CH, BATCH, wk_l, CH, 1, 0,
                    bufXSB, NS, 1, (long)CH * NS, 1.f, nullptr, nullptr, 0.f,
                    bk_l, 1, 0, bufKB, NS, 1, (long)CH * NS, 0);
        launch_gemm(stream, CH, NS, CH, BATCH, wv_l, CH, 1, 0,
                    bufXSA, NS, 1, (long)CH * NS, 1.f, nullptr, nullptr, 0.f,
                    bv_l, 1, 0, bufVA, NS, 1, (long)CH * NS, 0);
        launch_gemm(stream, CH, NS, CH, BATCH, wv_l, CH, 1, 0,
                    bufXSB, NS, 1, (long)CH * NS, 1.f, nullptr, nullptr, 0.f,
                    bv_l, 1, 0, bufVB, NS, 1, (long)CH * NS, 0);

        // ---- attn[bh,m,n] = SCALE*(qa^T ka + qb^T kb) ----
        launch_gemm(stream, HWSZ, NS, DD, BHD,
                    bufQA, 1, HWSZ, (long)DD * HWSZ,
                    bufKA, NS, 1, (long)DD * NS, ATT_SCALE,
                    bufQB, bufKB, ATT_SCALE,
                    nullptr, 0, 0,
                    bufATT, NS, 1, (long)HWSZ * NS, 0);

        // ---- RPE bias + softmax ----
        softmax_bias_kernel<<<BHD * HWSZ, 256, 0, stream>>>(bufATT, bufOfA, rpe_l);

        // ---- o[m,bh,d] = attn @ V^T  (writes into bufQA/bufQB, q is dead) ----
        launch_gemm(stream, HWSZ, DD, NS, BHD,
                    bufATT, NS, 1, (long)HWSZ * NS,
                    bufVA, 1, NS, (long)DD * NS, 1.f,
                    nullptr, nullptr, 0.f, nullptr, 0, 0,
                    bufQA, (long)BHD * DD, 1, DD, 0);
        launch_gemm(stream, HWSZ, DD, NS, BHD,
                    bufATT, NS, 1, (long)HWSZ * NS,
                    bufVB, 1, NS, (long)DD * NS, 1.f,
                    nullptr, nullptr, 0.f, nullptr, 0, 0,
                    bufQB, (long)BHD * DD, 1, DD, 0);

        // ---- cln1: x2 = residual(x) + CLN(o)  (attn region becomes x2) ----
        cln_kernel<<<HWSZ * BHD, 128, 0, stream>>>(
            bufQA, bufQB, xa, xb, 1,
            ln_grr + ((size_t)l * 2 + 0) * DD, ln_gri + ((size_t)l * 2 + 0) * DD,
            ln_gii + ((size_t)l * 2 + 0) * DD, ln_brA + ((size_t)l * 2 + 0) * DD,
            ln_biA + ((size_t)l * 2 + 0) * DD,
            xa2p, xb2p, 0);

        // ---- fc1 (complex linear + relu) ----
        launch_gemm(stream, HWSZ * BHD, DD, DD, 1,
                    xa2p, DD, 1, 0,
                    fc1_wr + (size_t)l * DD * DD, 1, DD, 0, 1.f,
                    xb2p, fc1_wi + (size_t)l * DD * DD, -1.f,
                    fc1_br + (size_t)l * DD, 0, 1,
                    bufHA, DD, 1, 0, 1);
        launch_gemm(stream, HWSZ * BHD, DD, DD, 1,
                    xa2p, DD, 1, 0,
                    fc1_wi + (size_t)l * DD * DD, 1, DD, 0, 1.f,
                    xb2p, fc1_wr + (size_t)l * DD * DD, 1.f,
                    fc1_bi + (size_t)l * DD, 0, 1,
                    bufHB, DD, 1, 0, 1);

        // ---- fc2 (complex linear) -> bufQA/bufQB ----
        launch_gemm(stream, HWSZ * BHD, DD, DD, 1,
                    bufHA, DD, 1, 0,
                    fc2_wr + (size_t)l * DD * DD, 1, DD, 0, 1.f,
                    bufHB, fc2_wi + (size_t)l * DD * DD, -1.f,
                    fc2_br + (size_t)l * DD, 0, 1,
                    bufQA, DD, 1, 0, 0);
        launch_gemm(stream, HWSZ * BHD, DD, DD, 1,
                    bufHA, DD, 1, 0,
                    fc2_wi + (size_t)l * DD * DD, 1, DD, 0, 1.f,
                    bufHB, fc2_wr + (size_t)l * DD * DD, 1.f,
                    fc2_bi + (size_t)l * DD, 0, 1,
                    bufQB, DD, 1, 0, 0);

        // ---- cln2: x3 = x2 + CLN(h2), written in [B,C,H,W] layout ----
        cln_kernel<<<HWSZ * BHD, 128, 0, stream>>>(
            bufQA, bufQB, xa2p, xb2p, 0,
            ln_grr + ((size_t)l * 2 + 1) * DD, ln_gri + ((size_t)l * 2 + 1) * DD,
            ln_gii + ((size_t)l * 2 + 1) * DD, ln_brA + ((size_t)l * 2 + 1) * DD,
            ln_biA + ((size_t)l * 2 + 1) * DD,
            outA, outB, 1);
    }
}
